// AttentionScore_causal_79860621902267
// MI455X (gfx1250) — matrix operand order repaired and checked by reading, not compiled, not run on hardware
//
#include <hip/hip_runtime.h>
#include <hip/hip_bf16.h>
#include <math.h>

// ---------------------------------------------------------------------------
// AttentionScore_causal for MI455X (gfx1250, wave32, WMMA bf16 path)
// B=8, S=2048, H=512
//   Wq,Wk -> bf16 once (workspace); qp = q @ Wq^T + bq ; kp = k @ Wk^T + bk
//   scores = (qp @ kp^T) * H^-0.5 * qc_score       (v_wmma_f32_16x16x32_bf16)
//   causal + key-padding mask -> -inf ; softmax rows -> d_out [B,S,S] f32
//
// Round-3 -> 4: the async-LDS builtin exists but takes int4-typed AS1/AS3
// pointers ("__device__ int4*" in the HIP diagnostic) — cast accordingly.
// ---------------------------------------------------------------------------

#define BATCH 8
#define SEQ   2048
#define HID   512
#define KSTEPS (HID / 32)        // 16 WMMA k-steps of K=32
#define ASTRIDE 520              // LDS row stride (bf16): 260 dwords -> row r
                                 // starts at bank (4*r)%64: conflict-free b128

typedef __attribute__((ext_vector_type(16))) __bf16 v16bf;
typedef __attribute__((ext_vector_type(8)))  __bf16 v8bf;
typedef __attribute__((ext_vector_type(4)))  __bf16 v4bf;
typedef __attribute__((ext_vector_type(8)))  float  v8f;
typedef __attribute__((ext_vector_type(4)))  float  v4f;
typedef __attribute__((ext_vector_type(4)))  int    v4i;

typedef __attribute__((address_space(1))) v4i* gptr_v4i;   // global int4*
typedef __attribute__((address_space(3))) v4i* lptr_v4i;   // LDS int4*

#if __has_builtin(__builtin_amdgcn_global_load_async_to_lds_b128)
#define HAVE_ASYNC_LDS 1
#else
#define HAVE_ASYNC_LDS 0
#endif

__device__ inline void wait_asynccnt0() {
#if __has_builtin(__builtin_amdgcn_s_wait_asynccnt)
    __builtin_amdgcn_s_wait_asynccnt(0);
#else
    asm volatile("s_wait_asynccnt 0x0" ::: "memory");
#endif
}

// A-matrix (16-bit, 16x32) fragment read from the LDS-staged tile.
// Lanes 0-15 hold K kk+0..15, lanes 16-31 hold K kk+16..31 (contiguous,
// matching the B fragment).
__device__ inline v16bf lds_a_frag(const __bf16* As, int l16, int kk, int half) {
    const __bf16* p = As + l16 * ASTRIDE + kk + half * 16;
    v8bf lo = *(const v8bf*)(p);          // ds_load_b128
    v8bf hi = *(const v8bf*)(p + 8);      // ds_load_b128
    return __builtin_shufflevector(lo, hi, 0, 1, 2, 3, 4, 5, 6, 7,
                                           8, 9, 10, 11, 12, 13, 14, 15);
}

// B-matrix (16-bit, 32x16) fragment, native bf16 source ([N,K] row-major, so
// column n of B == row n, K-contiguous). Lanes 0-15: K kk+0..15, lanes
// 16-31: K kk+16..31 (16 consecutive K per lane; one 32B read -> 2x b128).
__device__ inline v16bf load_b_bf16(const __bf16* __restrict__ row, int kk, int half) {
    return *(const v16bf*)(row + kk + half * 16);
}

// ---------------------------------------------------------------------------
// Kernel 0: elementwise f32 -> bf16 (used for Wq, Wk). n multiple of 2048.
// ---------------------------------------------------------------------------
__global__ __launch_bounds__(256) void
cvt_bf16_kernel(const float* __restrict__ X, __bf16* __restrict__ Y) {
    const int i = (blockIdx.x * 256 + threadIdx.x) * 8;
    v4f a = *(const v4f*)(X + i);
    v4f b = *(const v4f*)(X + i + 4);
    v8bf h;
#pragma unroll
    for (int j = 0; j < 4; ++j) { h[j] = (__bf16)a[j]; h[4 + j] = (__bf16)b[j]; }
    *(v8bf*)(Y + i) = h;
}

// ---------------------------------------------------------------------------
// Kernel 1: Y[M,512](bf16) = X[M,512](f32) @ Wbf[512,512]^T (bf16) + bias
// grid.x = M/16 ; 256 threads (8 waves). The 16x512 A tile is converted to
// bf16 and staged in LDS once; wave w owns N-tiles 4w..4w+3 (64 columns)
// with 4 accumulators, reading the shared A fragment once per k-step.
// ---------------------------------------------------------------------------
__global__ __launch_bounds__(256) void
proj_bf16_kernel(const float* __restrict__ X,
                 const __bf16* __restrict__ Wbf,
                 const float* __restrict__ bias,
                 __bf16* __restrict__ Y) {
    extern __shared__ __bf16 As[];            // [16][ASTRIDE]
    const int tid  = threadIdx.x;
    const int lane = tid & 31;
    const int wave = tid >> 5;
    const int half = lane >> 4;
    const int l16  = lane & 15;
    const int row0 = blockIdx.x * 16;

    // Cooperative stage: 16x512 f32 -> bf16 LDS (v4f loads, v4bf stores).
#pragma unroll
    for (int j = 0; j < 8; ++j) {
        const int c   = tid + j * 256;        // 2048 chunks of 4 elems
        const int row = c >> 7;               // 128 chunks per row
        const int col = (c & 127) * 4;
        v4f x = *(const v4f*)(X + (size_t)(row0 + row) * HID + col);
        v4bf h;
#pragma unroll
        for (int i = 0; i < 4; ++i) h[i] = (__bf16)x[i];
        *(v4bf*)(As + row * ASTRIDE + col) = h;
    }
    __syncthreads();

    v8f acc[4] = {};
    const __bf16* wbase = Wbf + (size_t)(wave * 64 + l16) * HID;
#pragma unroll
    for (int t = 0; t < KSTEPS; ++t) {
        const int kk = t * 32;
        v16bf a = lds_a_frag(As, l16, kk, half);
#pragma unroll
        for (int j = 0; j < 4; ++j) {
            v16bf b = load_b_bf16(wbase + (size_t)j * 16 * HID, kk, half);
            acc[j] = __builtin_amdgcn_wmma_f32_16x16x32_bf16(
                false, a, false, b, (short)0, acc[j], false, false);
        }
    }

#pragma unroll
    for (int j = 0; j < 4; ++j) {
        const int   col = wave * 64 + j * 16 + l16;
        const float bv  = bias[col];
#pragma unroll
        for (int i = 0; i < 8; ++i) {         // C/D: VGPR i -> M=i (+8 hi half)
            const int row = row0 + i + 8 * half;
            Y[(size_t)row * HID + col] = (__bf16)(acc[j][i] + bv);
        }
    }
}

// ---------------------------------------------------------------------------
// Kernel 2: scores + mask + softmax for one (batch, 16-row block).
// grid = (S/16, B), 256 threads. Dynamic LDS: 128 KB score tile + 16.3 KB
// staged qp block. Only causally-live N-tiles (n <= br) get WMMA; the dead
// right region of each output row is written as zeros directly.
// ---------------------------------------------------------------------------
__global__ __launch_bounds__(256) void
attn_softmax_kernel(const __bf16* __restrict__ qp,
                    const __bf16* __restrict__ kp,
                    const float* __restrict__ qc,
                    const unsigned char* __restrict__ kpm,
                    float* __restrict__ out) {
    extern __shared__ char smem[];
    float*  sc = (float*)smem;                          // [16][SEQ]
    __bf16* As = (__bf16*)(smem + 16 * SEQ * sizeof(float));  // [16][ASTRIDE]

    const int tid  = threadIdx.x;
    const int lane = tid & 31;
    const int wave = tid >> 5;
    const int half = lane >> 4;
    const int l16  = lane & 15;

    const int br = blockIdx.x;                // 16-row block index
    const int b  = blockIdx.y;
    const int s0 = br * 16;
    const int nActiveTiles = br + 1;
    const int nActive = nActiveTiles * 16;

    const __bf16* qpb = qp + ((size_t)b * SEQ + s0) * HID;
    const __bf16* kpb = kp + (size_t)b * SEQ * HID;
    const float scaling = 0.044194173824159216f;   // 512^-0.5

    // Stage this block's 16 query rows (bf16) into LDS: async DMA path on
    // CDNA5 when available (GLOBAL_LOAD_ASYNC_TO_LDS_B128, ASYNCcnt).
#pragma unroll
    for (int j = 0; j < 4; ++j) {
        const int c    = tid + j * 256;       // 1024 chunks of 8 bf16
        const int row  = c >> 6;              // 64 chunks per row
        const int col8 = (c & 63) * 8;
        const __bf16* g = qpb + (size_t)row * HID + col8;
        __bf16*       l = As + row * ASTRIDE + col8;
#if HAVE_ASYNC_LDS
        __builtin_amdgcn_global_load_async_to_lds_b128(
            (gptr_v4i)g, (lptr_v4i)l, 0, 0);
#else
        *(v8bf*)l = *(const v8bf*)g;
#endif
    }
#if HAVE_ASYNC_LDS
    wait_asynccnt0();
#endif
    __syncthreads();

    // Live score tiles, round-robin over the 8 waves (wave-uniform bounds,
    // EXEC stays all-ones around every WMMA).
    for (int n = wave; n < nActiveTiles; n += 8) {
        v8f acc = {};
        const __bf16* brow = kpb + (size_t)(n * 16 + l16) * HID;
#pragma unroll
        for (int t = 0; t < KSTEPS; ++t) {
            v16bf a = lds_a_frag(As, l16, t * 32, half);
            v16bf bb = load_b_bf16(brow, t * 32, half);
            acc = __builtin_amdgcn_wmma_f32_16x16x32_bf16(
                false, a, false, bb, (short)0, acc, false, false);
        }
        const int  tcol = n * 16 + l16;
        const bool pad  = kpm[(size_t)b * SEQ + tcol] != 0;
#pragma unroll
        for (int i = 0; i < 8; ++i) {
            const int row  = i + 8 * half;
            const int srow = s0 + row;
            float v = acc[i] * scaling * qc[((size_t)b * SEQ + srow) * SEQ + tcol];
            if (tcol > srow || pad) v = -INFINITY;
            sc[row * SEQ + tcol] = v;
        }
    }
    __syncthreads();

    // Row softmax: wave w handles rows 2w and 2w+1 (wave32 shuffle reduce).
#pragma unroll
    for (int rr = 0; rr < 2; ++rr) {
        const int r    = 2 * wave + rr;
        const int srow = s0 + r;
        float* rowp = sc + r * SEQ;

        float m = -INFINITY;
        for (int c = lane; c < nActive; c += 32) m = fmaxf(m, rowp[c]);
#pragma unroll
        for (int off = 16; off > 0; off >>= 1) m = fmaxf(m, __shfl_xor(m, off, 32));

        float ssum = 0.0f;
        for (int c = lane; c < nActive; c += 32) {
            float e = __expf(rowp[c] - m);
            rowp[c] = e;
            ssum += e;
        }
#pragma unroll
        for (int off = 16; off > 0; off >>= 1) ssum += __shfl_xor(ssum, off, 32);
        const float inv = 1.0f / ssum;

        float* op = out + ((size_t)b * SEQ + srow) * SEQ;
        for (int c = lane; c < nActive; c += 32) op[c] = rowp[c] * inv;
        for (int c = nActive + lane; c < SEQ; c += 32) op[c] = 0.0f;  // masked tail
    }
}

// ---------------------------------------------------------------------------
// Launch. Inputs (setup_inputs order):
//   0:q 1:k 2:attn_mask(bool, recomputed analytically) 3:key_padding_mask(bool)
//   4:qc_score 5:Wq 6:bq 7:Wk 8:bk
// Workspace: qp(16MB) | kp(16MB) | Wq_bf16(512KB) | Wk_bf16(512KB)
// ---------------------------------------------------------------------------
extern "C" void kernel_launch(void* const* d_in, const int* in_sizes, int n_in,
                              void* d_out, int out_size, void* d_ws, size_t ws_size,
                              hipStream_t stream) {
    const float* q   = (const float*)d_in[0];
    const float* k   = (const float*)d_in[1];
    const unsigned char* kpm = (const unsigned char*)d_in[3];
    const float* qc  = (const float*)d_in[4];
    const float* Wq  = (const float*)d_in[5];
    const float* bq  = (const float*)d_in[6];
    const float* Wk  = (const float*)d_in[7];
    const float* bk  = (const float*)d_in[8];
    float* out = (float*)d_out;

    const size_t projElems = (size_t)BATCH * SEQ * HID;   // 8M elems
    const size_t wElems    = (size_t)HID * HID;           // 256K elems
    __bf16* qp   = (__bf16*)d_ws;
    __bf16* kp   = qp + projElems;
    __bf16* WqBf = kp + projElems;
    __bf16* WkBf = WqBf + wElems;

    dim3 blk(256);
    cvt_bf16_kernel<<<dim3(wElems / (256 * 8)), blk, 0, stream>>>(Wq, WqBf);
    cvt_bf16_kernel<<<dim3(wElems / (256 * 8)), blk, 0, stream>>>(Wk, WkBf);

    const int M = BATCH * SEQ;                            // 16384 rows
    const size_t projLds = (size_t)16 * ASTRIDE * sizeof(__bf16);   // 16.3 KB
    proj_bf16_kernel<<<dim3(M / 16), blk, projLds, stream>>>(q, WqBf, bq, qp);
    proj_bf16_kernel<<<dim3(M / 16), blk, projLds, stream>>>(k, WkBf, bk, kp);

    const size_t attnLds = (size_t)16 * SEQ * sizeof(float) + projLds;  // ~147 KB
    attn_softmax_kernel<<<dim3(SEQ / 16, BATCH), blk, attnLds, stream>>>(
        qp, kp, qc, kpm, out);
}
